// MambaSequenceClassifierCNN_88459146428785
// MI455X (gfx1250) — matrix operand order repaired and checked by reading, not compiled, not run on hardware
//
#include <hip/hip_runtime.h>
#include <hip/hip_bf16.h>

// ---------------------------------------------------------------- constants
#define DMODEL   128
#define DINNER   256
#define DSTATE   16
#define DTRANK   8
#define NLAYERS  4
#define NCLS     100
#define BATCH    8
#define LIN      8192
#define L2C      4096     // after conv0 (stride 2)
#define L4C      2048     // after conv1 (stride 2)
#define TOK      (BATCH * L4C)   // 16384 tokens
#define XDP      48       // x_dbl width padded 40 -> 48 (3 WMMA tiles)
#define EPSV     1e-5f

typedef unsigned short ushort_t;
typedef __attribute__((ext_vector_type(16))) __bf16 bf16x16;
typedef __attribute__((ext_vector_type(8)))  float  f32x8;
typedef unsigned int u32x4 __attribute__((ext_vector_type(4)));
typedef int          i32x4 __attribute__((ext_vector_type(4)));
typedef int          i32x8 __attribute__((ext_vector_type(8)));

// f32 -> bf16 round-to-nearest-even (bit level; avoids relying on __bf16 arith)
__device__ __forceinline__ ushort_t f2bf(float f) {
  unsigned u = __builtin_bit_cast(unsigned, f);
  unsigned r = u + 0x7FFFu + ((u >> 16) & 1u);
  return (ushort_t)(r >> 16);
}

// ---------------------------------------------------------------- stem conv0
__global__ __launch_bounds__(256) void conv0_relu(
    const float* __restrict__ x, const float* __restrict__ w,
    const float* __restrict__ bias, float* __restrict__ out) {
  int idx = blockIdx.x * 256 + threadIdx.x;          // (b, co, l)
  int l  = idx & (L2C - 1);
  int co = (idx >> 12) & 127;
  int b  = idx >> 19;
  float acc = bias[co];
  int base = 2 * l - 4;
  for (int ci = 0; ci < 4; ++ci) {
    const float* xp = x + ((size_t)b * 4 + ci) * LIN;
    const float* wp = w + ((size_t)co * 4 + ci) * 9;
#pragma unroll
    for (int k = 0; k < 9; ++k) {
      int p = base + k;
      if ((unsigned)p < (unsigned)LIN) acc += xp[p] * wp[k];
    }
  }
  out[idx] = fmaxf(acc, 0.f);
}

// ---------------------------------------------------------------- stem conv1
__global__ __launch_bounds__(256) void conv1_relu(
    const float* __restrict__ h0, const float* __restrict__ w,
    const float* __restrict__ bias, float* __restrict__ out) {
  int idx = blockIdx.x * 256 + threadIdx.x;          // (b, co, l)
  int l  = idx & (L4C - 1);
  int co = (idx >> 11) & 127;
  int b  = idx >> 18;
  float acc = bias[co];
  int base = 2 * l - 4;
  const float* wp = w + (size_t)co * DMODEL * 9;
  const float* xp = h0 + (size_t)b * DMODEL * L2C;
  for (int ci = 0; ci < DMODEL; ++ci) {
    const float* xr = xp + (size_t)ci * L2C;
    const float* wr = wp + ci * 9;
#pragma unroll
    for (int k = 0; k < 9; ++k) {
      int p = base + k;
      if ((unsigned)p < (unsigned)L2C) acc += xr[p] * wr[k];
    }
  }
  out[idx] = fmaxf(acc, 0.f);
}

// -------------------------------------------------- batchnorm (train) helpers
__global__ __launch_bounds__(256) void bn_stats(
    const float* __restrict__ x, int lshift,
    float* __restrict__ mean, float* __restrict__ inv) {
  __shared__ float s1[256], s2[256];
  int c = blockIdx.x, tid = threadIdx.x;
  int Lc = 1 << lshift;
  int n = BATCH << lshift;
  float a = 0.f, q = 0.f;
  for (int i = tid; i < n; i += 256) {
    int bb = i >> lshift, l = i & (Lc - 1);
    float v = x[((size_t)bb * 128 + c) * Lc + l];
    a += v; q += v * v;
  }
  s1[tid] = a; s2[tid] = q; __syncthreads();
  for (int st = 128; st > 0; st >>= 1) {
    if (tid < st) { s1[tid] += s1[tid + st]; s2[tid] += s2[tid + st]; }
    __syncthreads();
  }
  if (tid == 0) {
    float mu = s1[0] / (float)n;
    float var = s2[0] / (float)n - mu * mu;
    mean[c] = mu;
    inv[c] = rsqrtf(var + EPSV);
  }
}

__global__ __launch_bounds__(256) void bn_apply(
    float* __restrict__ x, int lshift, const float* __restrict__ mean,
    const float* __restrict__ inv, const float* __restrict__ g,
    const float* __restrict__ b) {
  int idx = blockIdx.x * 256 + threadIdx.x;
  int c = (idx >> lshift) & 127;
  x[idx] = (x[idx] - mean[c]) * inv[c] * g[c] + b[c];
}

// ------------------------------------------------- NCH -> (token, channel)
__global__ __launch_bounds__(256) void nch_to_tc(
    const float* __restrict__ h1, float* __restrict__ h) {
  int idx = blockIdx.x * 256 + threadIdx.x;          // t*128 + c
  int c = idx & 127;
  int t = idx >> 7;
  int b = t >> 11;
  int l = t & (L4C - 1);
  h[idx] = h1[((size_t)b * 128 + c) * L4C + l];
}

// ------------------------------------------------- layernorm (+ bf16 emit)
__global__ __launch_bounds__(128) void ln128(
    const float* __restrict__ x, const float* __restrict__ g,
    const float* __restrict__ b, float* __restrict__ outf,
    ushort_t* __restrict__ outb) {
  __shared__ float s1[128], s2[128];
  int t = blockIdx.x, c = threadIdx.x;
  float v = x[(size_t)t * 128 + c];
  s1[c] = v; s2[c] = v * v; __syncthreads();
  for (int st = 64; st > 0; st >>= 1) {
    if (c < st) { s1[c] += s1[c + st]; s2[c] += s2[c + st]; }
    __syncthreads();
  }
  float mu = s1[0] * (1.f / 128.f);
  float var = s2[0] * (1.f / 128.f) - mu * mu;
  float o = (v - mu) * rsqrtf(var + EPSV) * g[c] + b[c];
  outf[(size_t)t * 128 + c] = o;
  outb[(size_t)t * 128 + c] = f2bf(o);
}

// ------------------------------------------------- weight f32 -> bf16 (+pad)
__global__ __launch_bounds__(256) void wcvt(
    const float* __restrict__ w, ushort_t* __restrict__ o,
    int Nact, int total, int K) {
  int idx = blockIdx.x * 256 + threadIdx.x;
  if (idx >= total) return;
  int n = idx / K;
  int k = idx - n * K;
  o[idx] = (n < Nact) ? f2bf(w[(size_t)n * K + k]) : (ushort_t)0;
}

// ------------------------------------------------- WMMA bf16 GEMM (global A)
// out[t, n] = sum_k A[t, k] * W[n, k];  K%32==0, N%16==0
__global__ __launch_bounds__(256) void gemm_bf16(
    const ushort_t* __restrict__ A, const ushort_t* __restrict__ W,
    float* __restrict__ out, int N, int K) {
  const int wave = blockIdx.x * 8 + (threadIdx.x >> 5);
  const int lane = threadIdx.x & 31;
  const int row  = lane & 15;
  const int hi   = lane >> 4;
  const int tilesN = N >> 4;
  const int tm = wave / tilesN;
  const int tn = wave - tm * tilesN;
  const int t0 = tm << 4;
  const int n0 = tn << 4;

  const ushort_t* ap = A + (size_t)(t0 + row) * K + hi * 8;
  const ushort_t* bp = W + (size_t)(n0 + row) * K + hi * 8;

  f32x8 acc = {};
  for (int kk = 0; kk < K; kk += 32) {
    union { bf16x16 v; uint4 q[2]; } af, bf;
    af.q[0] = *reinterpret_cast<const uint4*>(ap + kk);
    af.q[1] = *reinterpret_cast<const uint4*>(ap + kk + 16);
    bf.q[0] = *reinterpret_cast<const uint4*>(bp + kk);
    bf.q[1] = *reinterpret_cast<const uint4*>(bp + kk + 16);
    __builtin_prefetch(ap + kk + 32, 0, 0);
    acc = __builtin_amdgcn_wmma_f32_16x16x32_bf16(
        false, af.v, false, bf.v, (short)0, acc, false, false);
  }
  float* op = out + (size_t)(t0 + hi * 8) * N + n0 + row;
#pragma unroll
  for (int r = 0; r < 8; ++r) op[(size_t)r * N] = acc[r];
}

// --------------------------------- WMMA bf16 GEMM with TDM-staged A in LDS
// grid (T/16, N/128), 8 waves/block. A tile (16 x K bf16, <= 8KB) is DMA'd to
// LDS once by the Tensor Data Mover; all 8 waves (128 output cols) reuse it.
__global__ __launch_bounds__(256) void gemm_bf16_lds(
    const ushort_t* __restrict__ A, const ushort_t* __restrict__ W,
    float* __restrict__ out, int N, int K) {
  __shared__ __align__(16) ushort_t sA[16 * 256];   // LDS offset 0 (only LDS object)
  const int w    = threadIdx.x >> 5;
  const int lane = threadIdx.x & 31;
  const int row  = lane & 15;
  const int hi   = lane >> 4;
  const int t0   = blockIdx.x << 4;
  const int n0   = (blockIdx.y << 7) + (w << 4);

  // --- one wave programs the TDM with a 2D D# (ISA 8.3-8.5) and waits on it
  if (__builtin_amdgcn_readfirstlane(w) == 0) {
    unsigned long long ga =
        (unsigned long long)(size_t)(A + (size_t)t0 * K);
    u32x4 g0 = { 1u,                                   // count=1, no gather
                 0u,                                   // lds_addr = &sA = 0
                 (unsigned)ga,                         // global_addr[31:0]
                 ((unsigned)(ga >> 32) & 0x01FFFFFFu)  // global_addr[56:32]
                     | (2u << 30) };                   // type=2 ("image")
    i32x8 g1 = { (int)(1u << 16),                      // data_size=2B, mask=0
                 (int)(((unsigned)K & 0xFFFFu) << 16), // tensor_dim0 lo
                 (int)((((unsigned)K >> 16) & 0xFFFFu)
                     | (((unsigned)TOK & 0xFFFFu) << 16)), // dim0 hi | dim1 lo
                 (int)((((unsigned)TOK >> 16) & 0xFFFFu)
                     | ((unsigned)K << 16)),           // dim1 hi | tile_dim0=K
                 16,                                   // tile_dim1 = 16 rows
                 K,                                    // tensor_dim0_stride lo
                 0, 0 };
    i32x4 g2 = { 0, 0, 0, 0 };
    i32x4 g3 = { 0, 0, 0, 0 };
#if defined(__clang_major__) && (__clang_major__ >= 23)
    i32x8 g4 = { 0, 0, 0, 0, 0, 0, 0, 0 };
    __builtin_amdgcn_tensor_load_to_lds(g0, g1, g2, g3, g4, 0);
#else
    __builtin_amdgcn_tensor_load_to_lds(g0, g1, g2, g3, 0);
#endif
    __builtin_amdgcn_s_wait_tensorcnt(0);
  }
  __syncthreads();

  const ushort_t* aL = sA + row * K + hi * 8;         // ds_load_b128 path
  const ushort_t* bp = W + (size_t)(n0 + row) * K + hi * 8;

  f32x8 acc = {};
  for (int kk = 0; kk < K; kk += 32) {
    union { bf16x16 v; uint4 q[2]; } af, bf;
    af.q[0] = *reinterpret_cast<const uint4*>(aL + kk);
    af.q[1] = *reinterpret_cast<const uint4*>(aL + kk + 16);
    bf.q[0] = *reinterpret_cast<const uint4*>(bp + kk);
    bf.q[1] = *reinterpret_cast<const uint4*>(bp + kk + 16);
    acc = __builtin_amdgcn_wmma_f32_16x16x32_bf16(
        false, af.v, false, bf.v, (short)0, acc, false, false);
  }
  float* op = out + (size_t)(t0 + hi * 8) * N + n0 + row;
#pragma unroll
  for (int r = 0; r < 8; ++r) op[(size_t)r * N] = acc[r];
}

// --------------------------------- depthwise causal conv (D_CONV=4) + SiLU
__global__ __launch_bounds__(256) void mconv_silu(
    const float* __restrict__ xz, const float* __restrict__ cw,
    const float* __restrict__ cb, float* __restrict__ xm,
    ushort_t* __restrict__ xmb) {
  int idx = blockIdx.x * 256 + threadIdx.x;          // t*256 + d
  int d = idx & 255;
  int t = idx >> 8;
  int l = t & (L4C - 1);
  const float* wp = cw + d * 4;
  float acc = cb[d];
#pragma unroll
  for (int k = 0; k < 4; ++k) {
    int ll = l - 3 + k;
    if (ll >= 0) acc += xz[(size_t)(t - 3 + k) * 512 + d] * wp[k];
  }
  float sv = acc / (1.f + __expf(-acc));
  xm[idx] = sv;
  xmb[idx] = f2bf(sv);
}

// ------------------------------------------------- dt = softplus(dr@W^T + b)
__global__ __launch_bounds__(256) void dt_softplus(
    const float* __restrict__ xdbl, const float* __restrict__ dw,
    const float* __restrict__ db, float* __restrict__ dt) {
  int idx = blockIdx.x * 256 + threadIdx.x;          // t*256 + d
  int d = idx & 255;
  int t = idx >> 8;
  const float* xp = xdbl + (size_t)t * XDP;
  const float* wp = dw + d * 8;
  float a = db[d];
#pragma unroll
  for (int r = 0; r < 8; ++r) a += xp[r] * wp[r];
  dt[idx] = (a > 20.f) ? a : log1pf(__expf(a));
}

// ------------------------------------------------- selective scan
__global__ __launch_bounds__(256) void scan_kernel(
    const float* __restrict__ xm, const float* __restrict__ dt,
    const float* __restrict__ xdbl, const float* __restrict__ alog,
    const float* __restrict__ dskip, float* __restrict__ y) {
  __shared__ float bc[32];
  int b = blockIdx.x, d = threadIdx.x;
  float A[DSTATE];
#pragma unroll
  for (int n = 0; n < DSTATE; ++n) A[n] = -__expf(alog[d * DSTATE + n]);
  float Dp = dskip[d];
  float h[DSTATE];
#pragma unroll
  for (int n = 0; n < DSTATE; ++n) h[n] = 0.f;
  for (int l = 0; l < L4C; ++l) {
    int t = b * L4C + l;
    if (d < 32) bc[d] = xdbl[(size_t)t * XDP + 8 + d];   // B[0..15], C[0..15]
    __syncthreads();
    float dtv = dt[(size_t)t * 256 + d];
    float u   = xm[(size_t)t * 256 + d];
    float du  = dtv * u;
    float acc = 0.f;
#pragma unroll
    for (int n = 0; n < DSTATE; ++n) {
      h[n] = __expf(dtv * A[n]) * h[n] + du * bc[n];
      acc += h[n] * bc[16 + n];
    }
    y[(size_t)t * 256 + d] = acc + u * Dp;
    __syncthreads();
  }
}

// ------------------------------------------------- y * silu(z) -> bf16 act
__global__ __launch_bounds__(256) void gate_silu(
    const float* __restrict__ y, const float* __restrict__ xz,
    ushort_t* __restrict__ ob) {
  int idx = blockIdx.x * 256 + threadIdx.x;          // t*256 + d
  int d = idx & 255;
  int t = idx >> 8;
  float z = xz[(size_t)t * 512 + 256 + d];
  float v = y[idx] * (z / (1.f + __expf(-z)));
  ob[idx] = f2bf(v);
}

// ------------------------------------------------- residual add
__global__ __launch_bounds__(256) void resid_add(
    float* __restrict__ h, const float* __restrict__ o) {
  int idx = blockIdx.x * 256 + threadIdx.x;
  h[idx] += o[idx];
}

// ------------------------------------------------- mean pool over L
__global__ __launch_bounds__(128) void pool_kernel(
    const float* __restrict__ lnout, float* __restrict__ pooled) {
  int b = blockIdx.x, c = threadIdx.x;
  float s = 0.f;
  for (int l = 0; l < L4C; ++l) s += lnout[((size_t)b * L4C + l) * 128 + c];
  pooled[b * 128 + c] = s * (1.f / (float)L4C);
}

// ------------------------------------------------- classifier head
__global__ __launch_bounds__(128) void cls_kernel(
    const float* __restrict__ pooled, const float* __restrict__ w,
    const float* __restrict__ bias, float* __restrict__ out) {
  int b = blockIdx.x, j = threadIdx.x;
  if (j >= NCLS) return;
  const float* p = pooled + b * 128;
  const float* wr = w + j * 128;
  float a = bias[j];
  for (int c = 0; c < 128; ++c) a += p[c] * wr[c];
  out[b * NCLS + j] = a;
}

// ============================================================== launcher
extern "C" void kernel_launch(void* const* d_in, const int* in_sizes, int n_in,
                              void* d_out, int out_size, void* d_ws, size_t ws_size,
                              hipStream_t stream) {
  (void)in_sizes; (void)n_in; (void)out_size; (void)ws_size;
  const float* x        = (const float*)d_in[0];
  const float* conv0_w  = (const float*)d_in[1];
  const float* conv0_b  = (const float*)d_in[2];
  const float* bn0_g    = (const float*)d_in[3];
  const float* bn0_b    = (const float*)d_in[4];
  const float* conv1_w  = (const float*)d_in[5];
  const float* conv1_b  = (const float*)d_in[6];
  const float* bn1_g    = (const float*)d_in[7];
  const float* bn1_b    = (const float*)d_in[8];
  const float* ln_g     = (const float*)d_in[9];
  const float* ln_b     = (const float*)d_in[10];
  const float* in_proj  = (const float*)d_in[11];
  const float* mconv_w  = (const float*)d_in[12];
  const float* mconv_b  = (const float*)d_in[13];
  const float* xproj_w  = (const float*)d_in[14];
  const float* dtproj_w = (const float*)d_in[15];
  const float* dtproj_b = (const float*)d_in[16];
  const float* A_log    = (const float*)d_in[17];
  const float* D_skip   = (const float*)d_in[18];
  const float* outp_w   = (const float*)d_in[19];
  const float* fnorm_g  = (const float*)d_in[20];
  const float* fnorm_b  = (const float*)d_in[21];
  const float* cls_w    = (const float*)d_in[22];
  const float* cls_b    = (const float*)d_in[23];

  char* ws = (char*)d_ws;
  size_t off = 0;
  auto alloc = [&](size_t bytes) -> char* {
    char* p = ws + off;
    off = (off + bytes + 255) & ~(size_t)255;
    return p;
  };

  float*    XZ   = (float*)alloc((size_t)TOK * 512 * 4);
  float*    H0   = XZ;                         // stem-only aliases of XZ front
  float*    H1   = XZ + (size_t)8 * 128 * L2C;
  float*    H    = (float*)alloc((size_t)TOK * 128 * 4);
  float*    LN   = (float*)alloc((size_t)TOK * 128 * 4);
  ushort_t* ACT  = (ushort_t*)alloc((size_t)TOK * 256 * 2);
  float*    XM   = (float*)alloc((size_t)TOK * 256 * 4);
  float*    XDBL = (float*)alloc((size_t)TOK * XDP * 4);
  float*    DT   = (float*)alloc((size_t)TOK * 256 * 4);
  float*    Y    = (float*)alloc((size_t)TOK * 256 * 4);
  float*    OUTB = (float*)alloc((size_t)TOK * 128 * 4);
  ushort_t* WIN  = (ushort_t*)alloc((size_t)512 * 128 * 2);
  ushort_t* WXP  = (ushort_t*)alloc((size_t)XDP * 256 * 2);
  ushort_t* WOUT = (ushort_t*)alloc((size_t)128 * 256 * 2);
  float*    MEAN = (float*)alloc(128 * 4);
  float*    INV  = (float*)alloc(128 * 4);
  float*    POOL = (float*)alloc(8 * 128 * 4);

  // ---- stem
  conv0_relu<<<(8 * 128 * L2C) / 256, 256, 0, stream>>>(x, conv0_w, conv0_b, H0);
  bn_stats<<<128, 256, 0, stream>>>(H0, 12, MEAN, INV);
  bn_apply<<<(8 * 128 * L2C) / 256, 256, 0, stream>>>(H0, 12, MEAN, INV, bn0_g, bn0_b);
  conv1_relu<<<(8 * 128 * L4C) / 256, 256, 0, stream>>>(H0, conv1_w, conv1_b, H1);
  bn_stats<<<128, 256, 0, stream>>>(H1, 11, MEAN, INV);
  bn_apply<<<(8 * 128 * L4C) / 256, 256, 0, stream>>>(H1, 11, MEAN, INV, bn1_g, bn1_b);
  nch_to_tc<<<((size_t)TOK * 128) / 256, 256, 0, stream>>>(H1, H);

  // ---- Mamba layers
  for (int i = 0; i < NLAYERS; ++i) {
    wcvt<<<(512 * 128) / 256, 256, 0, stream>>>(
        in_proj + (size_t)i * 512 * 128, WIN, 512, 512 * 128, 128);
    wcvt<<<(XDP * 256) / 256, 256, 0, stream>>>(
        xproj_w + (size_t)i * 40 * 256, WXP, 40, XDP * 256, 256);
    wcvt<<<(128 * 256) / 256, 256, 0, stream>>>(
        outp_w + (size_t)i * 128 * 256, WOUT, 128, 128 * 256, 256);

    ln128<<<TOK, 128, 0, stream>>>(H, ln_g + i * 128, ln_b + i * 128, LN, ACT);
    // in_proj: (T,128) x (512,128)^T -> XZ (T,512)   [TDM-staged A]
    gemm_bf16_lds<<<dim3(TOK / 16, 512 / 128), 256, 0, stream>>>(ACT, WIN, XZ, 512, 128);
    mconv_silu<<<((size_t)TOK * 256) / 256, 256, 0, stream>>>(
        XZ, mconv_w + (size_t)i * 256 * 4, mconv_b + i * 256, XM, ACT);
    // x_proj: (T,256) x (48,256)^T -> XDBL (T,48)    [global A]
    gemm_bf16<<<(TOK / 16) * (XDP / 16) / 8, 256, 0, stream>>>(ACT, WXP, XDBL, XDP, 256);
    dt_softplus<<<((size_t)TOK * 256) / 256, 256, 0, stream>>>(
        XDBL, dtproj_w + (size_t)i * 256 * 8, dtproj_b + i * 256, DT);
    scan_kernel<<<BATCH, 256, 0, stream>>>(
        XM, DT, XDBL, A_log + (size_t)i * 256 * 16, D_skip + i * 256, Y);
    gate_silu<<<((size_t)TOK * 256) / 256, 256, 0, stream>>>(Y, XZ, ACT);
    // out_proj: (T,256) x (128,256)^T -> OUTB (T,128) [TDM-staged A]
    gemm_bf16_lds<<<dim3(TOK / 16, 128 / 128), 256, 0, stream>>>(ACT, WOUT, OUTB, 128, 256);
    resid_add<<<((size_t)TOK * 128) / 256, 256, 0, stream>>>(H, OUTB);
  }

  // ---- head
  ln128<<<TOK, 128, 0, stream>>>(H, fnorm_g, fnorm_b, LN, ACT);
  pool_kernel<<<BATCH, 128, 0, stream>>>(LN, POOL);
  cls_kernel<<<BATCH, 128, 0, stream>>>(POOL, cls_w, cls_b, (float*)d_out);
}
